// Net_50672024159110
// MI455X (gfx1250) — matrix operand order repaired and checked by reading, not compiled, not run on hardware
//
#include <hip/hip_runtime.h>
#include <hip/hip_bf16.h>

// ---------------------------------------------------------------------------
// LSTM (T=512, B=64, F=256, H=1024) + FC(1000) for gfx1250 (CDNA5, wave32).
// Persistent kernel, 64 blocks x 128 threads (256 waves). Block b owns hidden
// columns [16b, 16b+16); its FULL weight slice (4 gates x 16 cols x K=1280,
// bf16) = ~161 KB is staged into the 320 KB WGP LDS ONCE, so all 512 steps
// read the WMMA B operand via ds_load_b128. Cell state stays in VGPRs for the
// whole recurrence; accumulators start at the gate bias; steps separated by
// an agent-scope atomic generation barrier. K-loops use unroll-2 + manual
// double buffering (bounded live ranges: no spills, no vgpr-msb thrash).
// ---------------------------------------------------------------------------

typedef __bf16 bf16;
typedef bf16  v16bf __attribute__((ext_vector_type(16)));
typedef bf16  v8bf  __attribute__((ext_vector_type(8)));
typedef float v8f   __attribute__((ext_vector_type(8)));

#define T_STEPS 512
#define BATCH   64
#define FDIM    256
#define HDIM    1024
#define G4H     4096
#define KTOT    1280                 // FDIM + HDIM, concatenated in LDS
#define KPAD    1288                 // +8 elems => row stride 2576 B == 4 dwords
                                     // mod 64 banks: b128 hits each bank 2x (min)
#define NCLS    1000
#define NBLOCKS 64

union V16U { v16bf v; v8bf h[2]; };

// A operand (16x32 bf16; lane half hi holds k = 16*hi + 0..15). `p` includes row*lda + hi*16.
__device__ __forceinline__ v16bf a_frag(const bf16* __restrict__ p, int k0) {
  V16U u;
  u.h[0] = *reinterpret_cast<const v8bf*>(p + k0);
  u.h[1] = *reinterpret_cast<const v8bf*>(p + k0 + 8);
  return u.v;
}

// B operand (32x16 bf16): 32 contiguous bytes at `p + k0` (p includes hi*16).
__device__ __forceinline__ v16bf b_frag(const bf16* p, int k0) {
  const v8bf* q = reinterpret_cast<const v8bf*>(p + k0);
  V16U u;
  u.h[0] = q[0];
  u.h[1] = q[1];
  return u.v;
}

__device__ __forceinline__ float fast_sigmoid(float x) {
  return 1.0f / (1.0f + __expf(-x));
}
__device__ __forceinline__ float fast_tanh(float x) {
  x = fminf(15.0f, fmaxf(-15.0f, x));
  float e = __expf(2.0f * x);
  return (e - 1.0f) / (e + 1.0f);
}

// Agent-scope grid barrier (generation counter); all NBLOCKS blocks resident.
__device__ __forceinline__ void grid_barrier(unsigned* __restrict__ cnt,
                                             unsigned* __restrict__ gen,
                                             unsigned nblocks) {
  __builtin_amdgcn_fence(__ATOMIC_RELEASE, "agent");
  __syncthreads();
  if (threadIdx.x == 0) {
    unsigned g = __hip_atomic_load(gen, __ATOMIC_RELAXED, __HIP_MEMORY_SCOPE_AGENT);
    unsigned prev = __hip_atomic_fetch_add(cnt, 1u, __ATOMIC_ACQ_REL,
                                           __HIP_MEMORY_SCOPE_AGENT);
    if (prev == nblocks - 1u) {
      __hip_atomic_store(cnt, 0u, __ATOMIC_RELAXED, __HIP_MEMORY_SCOPE_AGENT);
      __hip_atomic_store(gen, g + 1u, __ATOMIC_RELEASE, __HIP_MEMORY_SCOPE_AGENT);
    } else {
      while (__hip_atomic_load(gen, __ATOMIC_ACQUIRE, __HIP_MEMORY_SCOPE_AGENT) == g) {
        __builtin_amdgcn_s_sleep(2);
      }
    }
  }
  __syncthreads();
  __builtin_amdgcn_fence(__ATOMIC_ACQUIRE, "agent");
}

// Pipelined K-loop (unroll 2): fragments for k+32 are loaded before the four
// WMMAs for k are issued, so waits before v_wmma are partial. Live ranges are
// bounded (~150 VGPRs): no scratch spills, no vgpr-msb switching.
__device__ __forceinline__ void gemm_phase(const bf16* __restrict__ ap, // A base (incl row*lda + hi*16)
                                           const bf16* b0, const bf16* b1,
                                           const bf16* b2, const bf16* b3,
                                           int kBase, int K, v8f* acc) {
  v16bf aC = a_frag(ap, 0);
  v16bf bC0 = b_frag(b0, kBase);
  v16bf bC1 = b_frag(b1, kBase);
  v16bf bC2 = b_frag(b2, kBase);
  v16bf bC3 = b_frag(b3, kBase);

#pragma unroll 2
  for (int k0 = 0; k0 < K; k0 += 32) {
    const int kn = (k0 + 32 < K) ? (k0 + 32) : k0;  // last iter: harmless reload
    v16bf aN  = a_frag(ap, kn);
    v16bf bN0 = b_frag(b0, kBase + kn);
    v16bf bN1 = b_frag(b1, kBase + kn);
    v16bf bN2 = b_frag(b2, kBase + kn);
    v16bf bN3 = b_frag(b3, kBase + kn);
    acc[0] = __builtin_amdgcn_wmma_f32_16x16x32_bf16(
        false, aC, false, bC0, (short)0, acc[0], false, false);
    acc[1] = __builtin_amdgcn_wmma_f32_16x16x32_bf16(
        false, aC, false, bC1, (short)0, acc[1], false, false);
    acc[2] = __builtin_amdgcn_wmma_f32_16x16x32_bf16(
        false, aC, false, bC2, (short)0, acc[2], false, false);
    acc[3] = __builtin_amdgcn_wmma_f32_16x16x32_bf16(
        false, aC, false, bC3, (short)0, acc[3], false, false);
    aC = aN;
    bC0 = bN0; bC1 = bN1; bC2 = bN2; bC3 = bN3;
  }
}

// ---------------------------------------------------------------------------
// Persistent LSTM. Block = ntile (blockIdx.x): hidden cols [16b,16b+16).
// Wave (0..3) = mtile: batch rows [16w,16w+16). B shared block-wide via LDS.
// ---------------------------------------------------------------------------
__global__ __launch_bounds__(128) void lstm_persistent_kernel(
    const bf16*  __restrict__ xall,   // [T, B, F] bf16 (preconverted)
    const float* __restrict__ Wih,    // [4H, F] fp32 (converted during staging)
    const float* __restrict__ Whh,    // [4H, H] fp32
    const float* __restrict__ b_ih,   // [4H]
    const float* __restrict__ b_hh,   // [4H]
    bf16*        __restrict__ hbuf0,  // [B, H] bf16 ping
    bf16*        __restrict__ hbuf1,  // [B, H] bf16 pong
    unsigned*    __restrict__ bar)    // [2]: counter, generation
{
  // Weight slice for this block: row r = gate(r>>4) x col(ntile*16 + (r&15)),
  // K = [W_ih row (256) | W_hh row (1024)] contiguous, bf16, stride KPAD.
  __shared__ bf16 sB[64 * KPAD];     // 164,864 B of the 320 KB WGP LDS

  const int lane   = threadIdx.x & 31;
  const int lane16 = lane & 15;
  const int hi     = lane >> 4;
  const int ntile  = blockIdx.x;           // 0..63
  const int mtile  = threadIdx.x >> 5;     // 0..3
  const int mbase  = mtile * 16;
  const int bcol   = ntile * 16 + lane16;  // this lane's hidden column
  const int arow   = mbase + lane16;       // this lane's batch row (A operand)

  // ---- Stage + convert the weight slice into LDS (once) ----
  // 64 rows x 160 chunks of 8 elements (1280 elems/row).
  for (int chunk = threadIdx.x; chunk < 64 * 160; chunk += 128) {
    const int row = chunk / 160;
    const int c8  = chunk - row * 160;
    const int k   = c8 * 8;
    const int q   = row >> 4;
    const int col = ntile * 16 + (row & 15);
    const float* src = (k < FDIM)
        ? (Wih + (size_t)(q * HDIM + col) * FDIM + k)
        : (Whh + (size_t)(q * HDIM + col) * HDIM + (k - FDIM));
    const float4* s4 = reinterpret_cast<const float4*>(src);
    float4 f0 = s4[0], f1 = s4[1];
    v8bf o;
    o[0] = (bf16)f0.x; o[1] = (bf16)f0.y; o[2] = (bf16)f0.z; o[3] = (bf16)f0.w;
    o[4] = (bf16)f1.x; o[5] = (bf16)f1.y; o[6] = (bf16)f1.z; o[7] = (bf16)f1.w;
    *reinterpret_cast<v8bf*>(&sB[(size_t)row * KPAD + k]) = o;
  }
  __syncthreads();

  // Independent per-gate LDS base pointers (row = q*16 + lane16).
  const bf16* b0 = &sB[(size_t)( 0 + lane16) * KPAD + hi * 16];
  const bf16* b1 = &sB[(size_t)(16 + lane16) * KPAD + hi * 16];
  const bf16* b2 = &sB[(size_t)(32 + lane16) * KPAD + hi * 16];
  const bf16* b3 = &sB[(size_t)(48 + lane16) * KPAD + hi * 16];

  // Loop-invariant biases (one scalar per gate per lane).
  const float bi = b_ih[0 * HDIM + bcol] + b_hh[0 * HDIM + bcol];
  const float bf = b_ih[1 * HDIM + bcol] + b_hh[1 * HDIM + bcol];
  const float bg = b_ih[2 * HDIM + bcol] + b_hh[2 * HDIM + bcol];
  const float bo = b_ih[3 * HDIM + bcol] + b_hh[3 * HDIM + bcol];

  // Register-resident cell state for the entire recurrence.
  float creg[8];
#pragma unroll
  for (int r = 0; r < 8; ++r) creg[r] = 0.0f;

  for (int t = 0; t < T_STEPS; ++t) {
    const bf16* hin  = (t & 1) ? hbuf1 : hbuf0;
    bf16*       hout = (t & 1) ? hbuf0 : hbuf1;
    const bf16* apx  = xall + (size_t)t * (BATCH * FDIM)
                            + (size_t)arow * FDIM + hi * 16;
    const bf16* aph  = hin + (size_t)arow * HDIM + hi * 16;

    // Accumulators start at the gate bias (saves the post-GEMM bias add).
    v8f acc[4];
#pragma unroll
    for (int r = 0; r < 8; ++r) {
      acc[0][r] = bi;
      acc[1][r] = bf;
      acc[2][r] = bg;
      acc[3][r] = bo;
    }

    gemm_phase(apx, b0, b1, b2, b3, 0,    FDIM, acc);  // x_t @ W_ih^T
    gemm_phase(aph, b0, b1, b2, b3, FDIM, HDIM, acc);  // h   @ W_hh^T

    // C/D layout: VGPR r -> row mbase + r + hi*8, col = bcol. Gates i,f,g,o.
#pragma unroll
    for (int r = 0; r < 8; ++r) {
      const int row = mbase + r + hi * 8;
      const float ig = fast_sigmoid(acc[0][r]);
      const float fg = fast_sigmoid(acc[1][r]);
      const float gg = fast_tanh  (acc[2][r]);
      const float og = fast_sigmoid(acc[3][r]);
      const float cn = fg * creg[r] + ig * gg;
      creg[r] = cn;
      hout[row * HDIM + bcol] = (bf16)(og * fast_tanh(cn));
    }

    grid_barrier(bar, bar + 1, (unsigned)NBLOCKS);
  }
}

// ---------------------------------------------------------------------------
// Final FC: out[64,1000] = h_last @ W_fc^T, K=1024. One-shot, 252 waves.
// ---------------------------------------------------------------------------
__global__ __launch_bounds__(128) void fc_kernel(
    const bf16*  __restrict__ h,    // [B, H] bf16
    const float* __restrict__ Wfc,  // [NCLS, H] fp32
    float*       __restrict__ out)  // [B, NCLS]
{
  const int lane   = threadIdx.x & 31;
  const int lane16 = lane & 15;
  const int hi     = lane >> 4;
  const int wid    = (blockIdx.x * blockDim.x + threadIdx.x) >> 5;  // 0..251
  const int ntile  = wid >> 2;   // 0..62
  const int mtile  = wid & 3;    // 0..3
  const int mbase  = mtile * 16;

  const int nraw = ntile * 16 + lane16;
  const int n    = nraw < NCLS ? nraw : (NCLS - 1);  // clamped addr, no divergence

  v8f acc;
#pragma unroll
  for (int r = 0; r < 8; ++r) acc[r] = 0.0f;

  const bf16*  ap = h + (size_t)(mbase + lane16) * HDIM + hi * 16;
  const float* wp = Wfc + (size_t)n * HDIM + hi * 16;

#pragma unroll 4
  for (int k0 = 0; k0 < HDIM; k0 += 32) {
    v16bf a = a_frag(ap, k0);
    const float4* fp = reinterpret_cast<const float4*>(wp + k0);
    float4 f0 = fp[0], f1 = fp[1], f2 = fp[2], f3 = fp[3];
    V16U u;
    u.v[0]  = (bf16)f0.x; u.v[1]  = (bf16)f0.y; u.v[2]  = (bf16)f0.z; u.v[3]  = (bf16)f0.w;
    u.v[4]  = (bf16)f1.x; u.v[5]  = (bf16)f1.y; u.v[6]  = (bf16)f1.z; u.v[7]  = (bf16)f1.w;
    u.v[8]  = (bf16)f2.x; u.v[9]  = (bf16)f2.y; u.v[10] = (bf16)f2.z; u.v[11] = (bf16)f2.w;
    u.v[12] = (bf16)f3.x; u.v[13] = (bf16)f3.y; u.v[14] = (bf16)f3.z; u.v[15] = (bf16)f3.w;
    acc = __builtin_amdgcn_wmma_f32_16x16x32_bf16(
        false, a, false, u.v, (short)0, acc, false, false);
  }

  if (nraw < NCLS) {
#pragma unroll
    for (int r = 0; r < 8; ++r) {
      const int row = mbase + r + hi * 8;
      out[row * NCLS + nraw] = acc[r];
    }
  }
}

// ---------------------------------------------------------------------------
// fp32 -> bf16 bulk conversion for x (8 elems/thread; size divisible by 8).
// ---------------------------------------------------------------------------
__global__ void f32_to_bf16_vec_kernel(const float* __restrict__ src,
                                       bf16* __restrict__ dst, int n8) {
  int i = blockIdx.x * blockDim.x + threadIdx.x;
  if (i < n8) {
    const float4* s = reinterpret_cast<const float4*>(src) + 2 * (size_t)i;
    float4 f0 = s[0], f1 = s[1];
    v8bf o;
    o[0] = (bf16)f0.x; o[1] = (bf16)f0.y; o[2] = (bf16)f0.z; o[3] = (bf16)f0.w;
    o[4] = (bf16)f1.x; o[5] = (bf16)f1.y; o[6] = (bf16)f1.z; o[7] = (bf16)f1.w;
    reinterpret_cast<v8bf*>(dst)[i] = o;
  }
}

__global__ void init_kernel(bf16* __restrict__ h0, unsigned* __restrict__ bar, int n) {
  int i = blockIdx.x * blockDim.x + threadIdx.x;
  if (i < n) h0[i] = (bf16)0.0f;
  if (i < 2) bar[i] = 0u;
}

// ---------------------------------------------------------------------------
extern "C" void kernel_launch(void* const* d_in, const int* in_sizes, int n_in,
                              void* d_out, int out_size, void* d_ws, size_t ws_size,
                              hipStream_t stream) {
  const float* x    = (const float*)d_in[0];  // [T, B, F]
  const float* W_ih = (const float*)d_in[1];  // [4H, F]
  const float* W_hh = (const float*)d_in[2];  // [4H, H]
  const float* b_ih = (const float*)d_in[3];  // [4H]
  const float* b_hh = (const float*)d_in[4];  // [4H]
  const float* W_fc = (const float*)d_in[5];  // [NCLS, H]
  float* out = (float*)d_out;
  (void)in_sizes; (void)n_in; (void)out_size; (void)ws_size;

  // Workspace carve-up (256B aligned).
  char* ws = (char*)d_ws;
  size_t off = 0;
  auto carve = [&](size_t bytes) {
    char* p = ws + off;
    off += (bytes + 255) & ~(size_t)255;
    return p;
  };
  bf16*     x_b   = (bf16*)carve((size_t)T_STEPS * BATCH * FDIM * sizeof(bf16)); // 16 MB
  bf16*     hbuf0 = (bf16*)carve((size_t)BATCH * HDIM * sizeof(bf16));
  bf16*     hbuf1 = (bf16*)carve((size_t)BATCH * HDIM * sizeof(bf16));
  unsigned* bar   = (unsigned*)carve(2 * sizeof(unsigned));

  // x conversion + state/barrier init (re-run every call: deterministic).
  {
    int nx8 = (T_STEPS * BATCH * FDIM) / 8;
    f32_to_bf16_vec_kernel<<<(nx8 + 255) / 256, 256, 0, stream>>>(x, x_b, nx8);
    int nh = BATCH * HDIM;
    init_kernel<<<(nh + 255) / 256, 256, 0, stream>>>(hbuf0, bar, nh);
  }

  // Entire 512-step recurrence in ONE persistent launch (weights -> LDS once).
  lstm_persistent_kernel<<<NBLOCKS, 128, 0, stream>>>(
      x_b, W_ih, W_hh, b_ih, b_hh, hbuf0, hbuf1, bar);

  // T even -> last h (written at t=511) lives in hbuf0.
  fc_kernel<<<63, 128, 0, stream>>>(hbuf0, W_fc, out);
}
